// MultiHeadGraphAttention_33131377722145
// MI455X (gfx1250) — matrix operand order repaired and checked by reading, not compiled, not run on hardware
//
#include <hip/hip_runtime.h>
#include <hip/hip_bf16.h>

// MultiHeadGraphAttention for MI455X (gfx1250), wave32 + WMMA bf16 pipeline.
//
// Sizes: N=4096, P=Q=128, H_DIM=256, HEADS=8, D=64.
// ws layout (bytes):
//   Wpack  [512][256][32] bf16   @ 0         (8,388,608)   W_bil re-tiled, B-fragment layout
//   Wtpack [8][512][32]   bf16   @ 8388608   (262,144)     Wt re-tiled, B-fragment layout
//   xpack  [8][4096][32]  bf16   @ 8650752   (2,097,152)   stage-1 out, A-fragment layout
//   xtB    [8][128][64][32] bf16 @ 10747904  (4,194,304)   stage-2 out, B-fragment layout (V)
//   s_src  [8][4096] f32         @ 14942208  (131,072)
//   s_dst  [8][4096] f32         @ 15073280  (131,072)
//   bmax   [8] f32               @ 15204352  (32)
// Total ~15.2 MB.

typedef __attribute__((ext_vector_type(16))) __bf16 v16bf;
typedef __attribute__((ext_vector_type(8)))  float  v8f;

union Frag {
    unsigned int u[8];
    uint4        q[2];
    v16bf        v;
};

__device__ __forceinline__ unsigned int pack2_bf16_trunc(float lo, float hi) {
    // D = { hi[31:16], lo[31:16] } in one v_perm_b32 (truncating f32->bf16)
    return __builtin_amdgcn_perm(__float_as_uint(hi), __float_as_uint(lo), 0x07060302u);
}
__device__ __forceinline__ unsigned short f32_to_bf16_rne(float f) {
    unsigned int u = __float_as_uint(f);
    u += 0x7fffu + ((u >> 16) & 1u);
    return (unsigned short)(u >> 16);
}
__device__ __forceinline__ float bf16_to_f32(unsigned short s) {
    return __uint_as_float(((unsigned int)s) << 16);
}
__device__ __forceinline__ float leaky(float t) { return t >= 0.f ? t : 0.2f * t; }

// packed bf16 multiply (VOP3P, ISA 15.10 op 42)
__device__ __forceinline__ unsigned int pk_mul_bf16(unsigned int a, unsigned int b) {
    unsigned int d;
    asm("v_pk_mul_bf16 %0, %1, %2" : "=v"(d) : "v"(a), "v"(b));
    return d;
}
// async DMA: 16B global -> LDS, tracked by ASYNCcnt (ISA 08_async_tensor §4)
__device__ __forceinline__ void async_copy_b128(unsigned ldsOff, const void* g) {
    asm volatile("global_load_async_to_lds_b128 %0, %1, off"
                 :: "v"(ldsOff), "v"((unsigned long long)(uintptr_t)g)
                 : "memory");
}

// ---------------------------------------------------------------------------
// K0a: W_bil [256][16384] f32  ->  Wpack [k/32][h][k%32] bf16
// ---------------------------------------------------------------------------
__global__ __launch_bounds__(256) void k0_pack_wbil(const float* __restrict__ W,
                                                    unsigned short* __restrict__ Wp) {
    int tid = blockIdx.x * 256 + threadIdx.x;     // 4,194,304 threads
    int h = tid >> 14;
    int k = tid & 16383;
    Wp[(size_t)(k >> 5) * (256u * 32u) + (size_t)h * 32u + (k & 31)] = f32_to_bf16_rne(W[tid]);
}

// ---------------------------------------------------------------------------
// K0b: Wt [512 cols][256 h] f32 -> Wtpack [h/32][col][h%32] bf16
// ---------------------------------------------------------------------------
__global__ __launch_bounds__(256) void k0_pack_wt(const float* __restrict__ Wt,
                                                  unsigned short* __restrict__ Wtp) {
    int tid = blockIdx.x * 256 + threadIdx.x;     // 131,072 threads
    int col = tid >> 8;
    int h   = tid & 255;
    Wtp[(size_t)(h >> 5) * (512u * 32u) + (size_t)col * 32u + (h & 31)] = f32_to_bf16_rne(Wt[tid]);
}

// ---------------------------------------------------------------------------
// K1: bilinear  x[n,h] = xp[n,:] W_bil[h] xn[n,:] + b_bil[h]
// GEMM M=4096, N=256, K=16384 with A synthesized from the rank-1 structure.
// Block: 256 thr = 8 waves, tile 64n x 256h. Wave: 16n (1 A-frag) x 128h
// (8 B-frags) -> 8 WMMA per K-chunk; the single A fragment (8 v_pk_mul_bf16
// per chunk) is reused across all 8 B fragments (~150 VGPRs, spill-free).
// B tiles (pair of K-chunks, 32KB, shared by all 4 n-groups) stream through a
// 3-slot LDS ring filled by global_load_async_to_lds_b128 with prefetch
// distance 2, gated by s_wait_asynccnt + one workgroup barrier per pair.
// The pair loop is unrolled x2 so qc = kc&3 is compile-time constant and the
// xnB fragment cache stays fully register-promoted (no v_movrels).
// ---------------------------------------------------------------------------
#define K1_NBUF 3
__global__ __launch_bounds__(256) void k1_bilinear(const float* __restrict__ xp,
                                                   const float* __restrict__ xn,
                                                   const unsigned short* __restrict__ Wp,
                                                   const float* __restrict__ bbil,
                                                   unsigned short* __restrict__ xpack) {
    __shared__ float xpT[64 * 128];
    __shared__ float xnT[64 * 128];
    __shared__ uint4 BtQ[K1_NBUF][2048];          // 3 x 32KB B-tile ring

    const int tid   = threadIdx.x;
    const int nbase = blockIdx.x * 64;
    #pragma unroll
    for (int i = 0; i < 32; ++i) {
        int idx = tid + i * 256;
        xpT[idx] = xp[(size_t)nbase * 128 + idx];
        xnT[idx] = xn[(size_t)nbase * 128 + idx];
    }

    // Each thread DMAs 128B of the 32KB pair-tile: 8 async b128 per pair.
    const char* wbase = (const char*)Wp;
    unsigned slotBase[K1_NBUF];
    #pragma unroll
    for (int s = 0; s < K1_NBUF; ++s)
        slotBase[s] = (unsigned)(uintptr_t)(&BtQ[s][0]);

    {   // prologue: pairs 0 and 1 in flight
        #pragma unroll
        for (int pr = 0; pr < 2; ++pr) {
            unsigned    dst = slotBase[pr] + tid * 128;
            const char* src = wbase + (size_t)pr * 32768u + tid * 128;
            #pragma unroll
            for (int j = 0; j < 8; ++j) async_copy_b128(dst + j * 16, src + j * 16);
        }
    }
    __syncthreads();    // xp/xn tiles ready for fragment pre-pack

    const int wave  = tid >> 5, lane = tid & 31;
    const int laneM = lane & 15;
    const int khalf = lane >> 4;
    const int kb    = khalf ? 8 : 0;      // A-layout K base per ISA table
    const int ngrp  = wave >> 1;          // 0..3 : 16-row group
    const int hb    = (wave & 1) * 128;   // 0 or 128
    const int row   = ngrp * 16 + laneM;

    // Pre-pack the xn fragments (4 q-subchunks) straight to bf16 pairs.
    unsigned xnB[4][8];
    #pragma unroll
    for (int qc = 0; qc < 4; ++qc)
        #pragma unroll
        for (int v = 0; v < 8; ++v) {
            int e0 = 2 * v, e1 = 2 * v + 1;
            int K0  = khalf * 16 + e0;
            int K1v = khalf * 16 + e1;
            xnB[qc][v] = pack2_bf16_trunc(xnT[row * 128 + qc * 32 + K0],
                                          xnT[row * 128 + qc * 32 + K1v]);
        }

    v8f acc[8] = {};
    #pragma unroll 2
    for (int pr = 0; pr < 256; ++pr) {            // pair = 2 K-chunks
        // own DMA of pair `pr` done (8 newer ops may stay in flight);
        // barrier then publishes every wave's slice.
        if (pr < 255) asm volatile("s_wait_asynccnt 0x8" ::: "memory");
        else          asm volatile("s_wait_asynccnt 0x0" ::: "memory");
        __syncthreads();
        if (pr + 2 < 256) {                        // prefetch pair pr+2
            unsigned    dst = slotBase[(pr + 2) % K1_NBUF] + tid * 128;
            const char* src = wbase + (size_t)(pr + 2) * 32768u + tid * 128;
            #pragma unroll
            for (int j = 0; j < 8; ++j) async_copy_b128(dst + j * 16, src + j * 16);
        }

        const unsigned short* bt = (const unsigned short*)&BtQ[pr % K1_NBUF][0];
        #pragma unroll
        for (int half = 0; half < 2; ++half) {
            int kc = pr * 2 + half;                // k = p*128+q ; kc = p*4 + q/32
            int p  = kc >> 2, qc = kc & 3;         // qc const after x2 unroll
            float    s  = xpT[row * 128 + p];
            unsigned sp = pack2_bf16_trunc(s, s);  // broadcast bf16 pair
            Frag a;
            #pragma unroll
            for (int v = 0; v < 8; ++v)
                a.u[v] = pk_mul_bf16(xnB[qc][v], sp);
            const unsigned short* brow = bt + half * 8192 + khalf * 16;
            #pragma unroll
            for (int f = 0; f < 8; ++f) {
                int h = hb + f * 16 + laneM;
                const uint4* bp = (const uint4*)(brow + (size_t)h * 32u);
                Frag b; b.q[0] = bp[0]; b.q[1] = bp[1];
                acc[f] = __builtin_amdgcn_wmma_f32_16x16x32_bf16(
                    false, a.v, false, b.v, (short)0, acc[f], false, false);
            }
        }
    }

    // Epilogue: + bias, convert to bf16, store in stage-2 A-fragment layout.
    #pragma unroll
    for (int f = 0; f < 8; ++f) {
        int   h    = hb + f * 16 + laneM;
        float bias = bbil[h];
        int   hc = h >> 5, hin = h & 31;
        unsigned short* dst = xpack + (size_t)hc * (4096u * 32u) + hin;
        #pragma unroll
        for (int r = 0; r < 8; ++r) {
            int n = nbase + ngrp * 16 + khalf * 8 + r;
            dst[(size_t)n * 32u] = f32_to_bf16_rne(acc[f][r] + bias);
        }
    }
}

// ---------------------------------------------------------------------------
// K2: xt = x @ Wt^T.  GEMM M=4096, cols=512 (k*64+d), K=256 (8 chunks).
// Writes xt as bf16 in stage-3 B-fragment layout xtB[k][jc][d][jin].
// ---------------------------------------------------------------------------
__global__ __launch_bounds__(256) void k2_proj(const unsigned short* __restrict__ xpack,
                                               const unsigned short* __restrict__ Wtp,
                                               unsigned short* __restrict__ xtB) {
    const int tid   = threadIdx.x;
    const int wave  = tid >> 5, lane = tid & 31;
    const int laneM = lane & 15;
    const int khalf = lane >> 4;
    const int nbase = blockIdx.x * 32;
    const int nb    = (wave >> 2) * 16;
    const int colb  = blockIdx.y * 256 + (wave & 3) * 64;
    const int row   = nbase + nb + laneM;

    v8f acc[4] = {};
    #pragma unroll 1
    for (int hc = 0; hc < 8; ++hc) {
        const unsigned short* ap = xpack + ((size_t)hc * 4096u + row) * 32u + (khalf ? 16 : 0);
        Frag a; a.q[0] = *(const uint4*)ap; a.q[1] = *(const uint4*)(ap + 8);
        const unsigned short* brow = Wtp + (size_t)hc * (512u * 32u) + (size_t)khalf * 16u;
        #pragma unroll
        for (int f = 0; f < 4; ++f) {
            int c = colb + f * 16 + laneM;
            const uint4* bp = (const uint4*)(brow + (size_t)c * 32u);
            Frag b; b.q[0] = bp[0]; b.q[1] = bp[1];
            acc[f] = __builtin_amdgcn_wmma_f32_16x16x32_bf16(
                false, a.v, false, b.v, (short)0, acc[f], false, false);
        }
    }

    const int jc   = nbase >> 5;
    const int jinb = nb + khalf * 8;
    #pragma unroll
    for (int f = 0; f < 4; ++f) {
        int c = colb + f * 16 + laneM;
        int k = c >> 6, d = c & 63;
        unsigned short* dst = xtB + (((size_t)k * 128u + jc) * 64u + d) * 32u + jinb;
        #pragma unroll
        for (int r = 0; r < 8; r += 2) {
            unsigned int w = (unsigned int)f32_to_bf16_rne(acc[f][r]) |
                             ((unsigned int)f32_to_bf16_rne(acc[f][r + 1]) << 16);
            *(unsigned int*)(dst + r) = w;
        }
    }
}

// ---------------------------------------------------------------------------
// K3: s_src[k,n] = xt[k,n,:].a_src[k];  s_dst likewise (a_vec[:,64:]).
// ---------------------------------------------------------------------------
__global__ __launch_bounds__(256) void k3_scores(const unsigned short* __restrict__ xtB,
                                                 const float* __restrict__ a_vec,
                                                 float* __restrict__ s_src,
                                                 float* __restrict__ s_dst) {
    int tid = blockIdx.x * 256 + threadIdx.x;     // 32768 = 8*4096
    int k = tid >> 12, n = tid & 4095;
    const unsigned short* base = xtB + (((size_t)k * 128u + (n >> 5)) * 64u) * 32u + (n & 31);
    float ss = 0.f, sd = 0.f;
    #pragma unroll 8
    for (int d = 0; d < 64; ++d) {
        float v = bf16_to_f32(base[(size_t)d * 32u]);
        ss += v * a_vec[k * 128 + d];
        sd += v * a_vec[k * 128 + 64 + d];
    }
    s_src[tid] = ss;
    s_dst[tid] = sd;
}

// ---------------------------------------------------------------------------
// K4: bmax[k] = max_j s_dst[k,j]  (leaky is monotone, so the softmax row max
// is leaky(a_i + bmax[k]) — one 8-value reduction replaces a full max pass)
// ---------------------------------------------------------------------------
__global__ __launch_bounds__(256) void k4_bmax(const float* __restrict__ s_dst,
                                               float* __restrict__ bmax) {
    __shared__ float red[256];
    int k = blockIdx.x;
    float m = -3.4e38f;
    for (int i = threadIdx.x; i < 4096; i += 256)
        m = fmaxf(m, s_dst[(size_t)k * 4096u + i]);
    red[threadIdx.x] = m;
    __syncthreads();
    for (int s = 128; s > 0; s >>= 1) {
        if (threadIdx.x < (unsigned)s) red[threadIdx.x] = fmaxf(red[threadIdx.x], red[threadIdx.x + s]);
        __syncthreads();
    }
    if (threadIdx.x == 0) bmax[k] = red[0];
}

// ---------------------------------------------------------------------------
// K5: flash-style attention. P-tile (16x32) synthesized in VALU from the
// rank-1 score structure, fed to WMMA against bf16 V-tiles. The s_dst row is
// staged to LDS via async DMA; V fragments run through a register double
// buffer so the ~100-instruction exp/leaky P build hides the VMEM latency.
// Block: 128 thr = 4 waves (4 i-tiles).
// ---------------------------------------------------------------------------
__global__ __launch_bounds__(128) void k5_attn(const unsigned short* __restrict__ xtB,
                                               const float* __restrict__ s_src,
                                               const float* __restrict__ s_dst,
                                               const float* __restrict__ bmax,
                                               float* __restrict__ out) {
    __shared__ uint4 bshQ[1024];                  // 16KB: s_dst row of this head
    __shared__ float zsh[4][16];
    const float* bsh = (const float*)&bshQ[0];
    const int k   = blockIdx.y;
    const int tid = threadIdx.x;

    {   // async-DMA the 16KB row: 128 thr x 8 b128
        unsigned    dst = (unsigned)(uintptr_t)(&bshQ[0]) + tid * 128;
        const char* src = (const char*)(s_dst + (size_t)k * 4096u) + tid * 128;
        #pragma unroll
        for (int j = 0; j < 8; ++j) async_copy_b128(dst + j * 16, src + j * 16);
        asm volatile("s_wait_asynccnt 0x0" ::: "memory");
    }
    __syncthreads();

    const int wave  = tid >> 5, lane = tid & 31;
    const int laneM = lane & 15;
    const int khalf = lane >> 4;
    const int kb    = khalf ? 8 : 0;
    const int ib    = blockIdx.x * 64 + wave * 16;

    const float arow = s_src[(size_t)k * 4096u + ib + laneM];
    const float m    = leaky(arow + bmax[k]);     // exact row max of leaky(a_i+b_j)

    v8f   acc[4] = {};
    float zacc   = 0.f;
    const unsigned short* vb = xtB + (size_t)k * (128u * 64u * 32u) + (size_t)khalf * 16u;

    Frag bb[2][4];
    #pragma unroll
    for (int f = 0; f < 4; ++f) {                 // preload jc=0
        const uint4* bp = (const uint4*)(vb + (size_t)(f * 16 + laneM) * 32u);
        bb[0][f].q[0] = bp[0]; bb[0][f].q[1] = bp[1];
    }

    #pragma unroll 1
    for (int jc = 0; jc < 128; ++jc) {
        const int cur = jc & 1;
        if (jc + 1 < 128) {                       // issue next V tile first...
            const unsigned short* brow = vb + (size_t)(jc + 1) * (64u * 32u);
            #pragma unroll
            for (int f = 0; f < 4; ++f) {
                const uint4* bp = (const uint4*)(brow + (size_t)(f * 16 + laneM) * 32u);
                bb[cur ^ 1][f].q[0] = bp[0]; bb[cur ^ 1][f].q[1] = bp[1];
            }
        }
        Frag a;                                   // ...then hide it under the P build
        #pragma unroll
        for (int v = 0; v < 8; ++v) {
            int e0 = 2 * v, e1 = 2 * v + 1;
            int K0 = khalf * 16 + e0;
            int K1 = khalf * 16 + e1;
            float p0 = __expf(leaky(arow + bsh[jc * 32 + K0]) - m);
            float p1 = __expf(leaky(arow + bsh[jc * 32 + K1]) - m);
            zacc += p0 + p1;
            a.u[v] = pack2_bf16_trunc(p0, p1);
        }
        #pragma unroll
        for (int f = 0; f < 4; ++f)
            acc[f] = __builtin_amdgcn_wmma_f32_16x16x32_bf16(
                false, a.v, false, bb[cur][f].v, (short)0, acc[f], false, false);
    }

    // Combine Z across the two K-halves (lanes l and l^16 hold the same row M).
    float Z = zacc + __shfl_xor(zacc, 16, 32);
    if (lane < 16) zsh[wave][laneM] = Z;
    __syncthreads();
    float zr[8];
    #pragma unroll
    for (int r = 0; r < 8; ++r) zr[r] = zsh[wave][kb + r];

    #pragma unroll
    for (int f = 0; f < 4; ++f) {
        int col = k * 64 + f * 16 + laneM;
        #pragma unroll
        for (int r = 0; r < 8; ++r) {
            int i = ib + kb + r;
            out[(size_t)i * 512u + col] = tanhf(acc[f][r] / zr[r]);
        }
    }
}

// ---------------------------------------------------------------------------
extern "C" void kernel_launch(void* const* d_in, const int* in_sizes, int n_in,
                              void* d_out, int out_size, void* d_ws, size_t ws_size,
                              hipStream_t stream) {
    const float* x_prices = (const float*)d_in[0];
    const float* x_news   = (const float*)d_in[1];
    const float* W_bil    = (const float*)d_in[2];
    const float* b_bil    = (const float*)d_in[3];
    const float* Wt       = (const float*)d_in[4];
    const float* a_vec    = (const float*)d_in[5];
    float* out = (float*)d_out;

    char* ws = (char*)d_ws;
    unsigned short* Wpack  = (unsigned short*)(ws + 0);
    unsigned short* Wtpack = (unsigned short*)(ws + 8388608);
    unsigned short* xpack  = (unsigned short*)(ws + 8650752);
    unsigned short* xtB    = (unsigned short*)(ws + 10747904);
    float*          s_src  = (float*)(ws + 14942208);
    float*          s_dst  = (float*)(ws + 15073280);
    float*          bmaxp  = (float*)(ws + 15204352);

    k0_pack_wbil<<<16384, 256, 0, stream>>>(W_bil, Wpack);
    k0_pack_wt  <<<512,   256, 0, stream>>>(Wt, Wtpack);
    k1_bilinear <<<64,    256, 0, stream>>>(x_prices, x_news, Wpack, b_bil, xpack);
    k2_proj     <<<dim3(128, 2), 256, 0, stream>>>(xpack, Wtpack, xtB);
    k3_scores   <<<128,   256, 0, stream>>>(xtB, a_vec, s_src, s_dst);
    k4_bmax     <<<8,     256, 0, stream>>>(s_dst, bmaxp);
    k5_attn     <<<dim3(64, 8), 128, 0, stream>>>(xtB, s_src, s_dst, bmaxp, out);

    (void)in_sizes; (void)n_in; (void)out_size; (void)ws_size;
}
